// GAT_4449586119269
// MI455X (gfx1250) — matrix operand order repaired and checked by reading, not compiled, not run on hardware
//
#include <hip/hip_runtime.h>

// ---------------- problem constants ----------------
#define N_NODES 8192
#define F_IN    512
#define F_HID   64
#define HEADS   4
#define CLASSES 20
#define SLOPE   0.2f
#define NEG_MASK_F (-90000000000.0f)
#define LOG2E   1.4426950408889634f

typedef __attribute__((ext_vector_type(16))) _Float16 v16h;
typedef __attribute__((ext_vector_type(8)))  float    v8f;

static __device__ inline v8f wmma_f16(v16h a, v16h b, v8f c) {
  // D = A(16x32 f16) * B(32x16 f16) + C(16x16 f32)
  return __builtin_amdgcn_wmma_f32_16x16x32_f16(
      /*neg_a=*/false, a, /*neg_b=*/false, b,
      /*c_mod=*/(short)0, c, /*reuse_a=*/false, /*reuse_b=*/false);
}

static __device__ inline v8f v8f_zero() {
  v8f z = {0.f,0.f,0.f,0.f,0.f,0.f,0.f,0.f};
  return z;
}

// ---------------- 1) adjacency -> bitmask (8192 x 256 u32, 8MB, L2-resident)
__global__ __launch_bounds__(256) void adjpack_kernel(const int* __restrict__ adj,
                                                      unsigned int* __restrict__ bits) {
  size_t w = (size_t)blockIdx.x * 256 + threadIdx.x;   // one u32 word per thread
  const int4* p = (const int4*)(adj + w * 32);
  unsigned int word = 0;
#pragma unroll
  for (int q = 0; q < 8; ++q) {
    int4 v = p[q];
    if (v.x > 0) word |= 1u << (q * 4 + 0);
    if (v.y > 0) word |= 1u << (q * 4 + 1);
    if (v.z > 0) word |= 1u << (q * 4 + 2);
    if (v.w > 0) word |= 1u << (q * 4 + 3);
  }
  bits[w] = word;
}

// ---------------- 2) pack f32 matrix [K][ld] into WMMA-B f16 fragment layout
// fragment (kb,n): lane l holds 16 halves, element e -> K=kb*32+(l<16?0:8)+e+(e<8?0:8),
// N = n*16 + (l&15).  Stored contiguously: ((kb*n_count+n)*32 + l)*16 + e.
// Columns >= Ncols are zero except column onesCol, which is the constant 1.0
// "denominator" column (softmax row-sum computed by the WMMA itself).
__global__ __launch_bounds__(256) void pack_b_kernel(const float* __restrict__ src,
                                                     _Float16* __restrict__ dst,
                                                     int ld, int Ncols, int n_count,
                                                     int onesCol,
                                                     size_t srcStride, size_t dstStride) {
  int tid  = blockIdx.x * 256 + threadIdx.x;
  int b    = blockIdx.y;
  int l    = tid & 31;
  int frag = tid >> 5;
  int n    = frag % n_count;
  int kb   = frag / n_count;
  const float* s = src + (size_t)b * srcStride;
  _Float16*    d = dst + (size_t)b * dstStride + ((size_t)(kb * n_count + n) * 32 + l) * 16;
  int col   = n * 16 + (l & 15);
  int kbase = kb * 32 + ((l < 16) ? 0 : 8);
#pragma unroll
  for (int e = 0; e < 16; ++e) {
    float v = (col < Ncols) ? s[(size_t)(kbase + e + ((e < 8) ? 0 : 8)) * ld + col]
                            : ((col == onesCol) ? 1.0f : 0.0f);
    d[e] = (_Float16)v;
  }
}

// ---------------- 3) layer-1 projection: Wh1[h] = x @ W1[h] via WMMA f16
__global__ __launch_bounds__(32) void proj1_kernel(const float* __restrict__ x,
                                                   const _Float16* __restrict__ packedW1,
                                                   float* __restrict__ Wh1) {
  const int l    = threadIdx.x;
  const int head = blockIdx.y;
  const int m0   = blockIdx.x * 16;
  const int ml   = l & 15;
  const int klo  = (l < 16) ? 0 : 8;
  const size_t row = (size_t)(m0 + ml);
  const _Float16* pW = packedW1 + (size_t)head * (16 * 4 * 512);

  v8f acc[4];
#pragma unroll
  for (int n = 0; n < 4; ++n) acc[n] = v8f_zero();

  for (int kt = 0; kt < F_IN / 32; ++kt) {
    const float* base = x + row * F_IN + kt * 32 + klo;   // 32B aligned
    float4 r0 = ((const float4*)base)[0];
    float4 r1 = ((const float4*)base)[1];
    float4 r2 = ((const float4*)base)[4];
    float4 r3 = ((const float4*)base)[5];
    v16h a;
    a[0]=(_Float16)r0.x; a[1]=(_Float16)r0.y; a[2]=(_Float16)r0.z; a[3]=(_Float16)r0.w;
    a[4]=(_Float16)r1.x; a[5]=(_Float16)r1.y; a[6]=(_Float16)r1.z; a[7]=(_Float16)r1.w;
    a[8]=(_Float16)r2.x; a[9]=(_Float16)r2.y; a[10]=(_Float16)r2.z; a[11]=(_Float16)r2.w;
    a[12]=(_Float16)r3.x; a[13]=(_Float16)r3.y; a[14]=(_Float16)r3.z; a[15]=(_Float16)r3.w;
#pragma unroll
    for (int n = 0; n < 4; ++n) {
      v16h b = *(const v16h*)(pW + ((size_t)(kt * 4 + n) * 32 + l) * 16);
      acc[n] = wmma_f16(a, b, acc[n]);
    }
  }
#pragma unroll
  for (int n = 0; n < 4; ++n) {
    int col = n * 16 + ml;
#pragma unroll
    for (int r = 0; r < 8; ++r) {
      int rowo = r + ((l < 16) ? 0 : 8);
      Wh1[((size_t)head * N_NODES + m0 + rowo) * F_HID + col] = acc[n][r];
    }
  }
}

// ---------------- 4) s,d vectors: s_i = Wh[i,:]@a_src, d_i = Wh[i,:]@a_dst
__global__ __launch_bounds__(256) void sd_kernel(const float* __restrict__ Wh,
                                                 const float* __restrict__ alpha,
                                                 float* __restrict__ S, float* __restrict__ D,
                                                 int F) {
  int row = blockIdx.x * 256 + threadIdx.x;
  int b   = blockIdx.y;
  const float* w  = Wh + ((size_t)b * N_NODES + row) * F;
  const float* as = alpha + (size_t)b * 2 * F;
  float s = 0.f, d = 0.f;
  for (int f = 0; f < F; ++f) { float v = w[f]; s += v * as[f]; d += v * as[F + f]; }
  S[(size_t)b * N_NODES + row] = s;
  D[(size_t)b * N_NODES + row] = d;
}

// ---------------- 5) per-row softmax max: M_i = max(lrelu(s_i + max_adj d_j), NEG_MASK)
__global__ __launch_bounds__(32) void rowmax_kernel(const unsigned int* __restrict__ bits,
                                                    const float* __restrict__ D,
                                                    const float* __restrict__ S,
                                                    float* __restrict__ M) {
  int row = blockIdx.x, b = blockIdx.y, l = threadIdx.x;
  const float* d = D + (size_t)b * N_NODES;
  float mx = -3.0e38f;
  for (int w = l; w < N_NODES / 32; w += 32) {
    unsigned int word = bits[(size_t)row * (N_NODES / 32) + w];
    while (word) {
      int bit = __ffs(word) - 1;
      word &= word - 1;
      mx = fmaxf(mx, d[w * 32 + bit]);
    }
  }
#pragma unroll
  for (int off = 16; off; off >>= 1) mx = fmaxf(mx, __shfl_xor(mx, off, 32));
  if (l == 0) {
    float e = S[(size_t)b * N_NODES + row] + mx;
    e = (e > 0.f) ? e : SLOPE * e;
    M[(size_t)b * N_NODES + row] = fmaxf(e, NEG_MASK_F);
  }
}

// ---------------- 6) fused masked-softmax attention aggregation (flash-style)
// out[i,:] = elu( (softmax_j mask(lrelu(s_i+d_j))) @ Wh )
// 8 waves/block, each owns one 16-row block; all share head + j-loop, so the
// packed f16 B tiles for 4 j-blocks (128 cols) are staged per barrier round.
// NC fragments per j-block; fragment DN column DLANE is the constant-1 column,
// so the softmax denominator falls out of the f32 WMMA accumulator.
template <int NC, int DN, int DLANE>
__global__ __launch_bounds__(256) void agg_kernel(const _Float16* __restrict__ packedB,
                                                  const float* __restrict__ S,
                                                  const float* __restrict__ D,
                                                  const float* __restrict__ M,
                                                  const unsigned int* __restrict__ adjbits,
                                                  float* __restrict__ out,
                                                  int colLimit, int outStride,
                                                  int headColMul, size_t packedHeadStride) {
  __shared__ __align__(16) _Float16 Btile[4 * NC * 512];   // 4 j-blocks staged per round
  __shared__ float dblk[128];
  const int tid  = threadIdx.x;
  const int wave = tid >> 5;
  const int l    = tid & 31;
  const int head = blockIdx.y;
  const int rowg = (blockIdx.x * 8 + wave) * 16;
  const int ml   = l & 15;
  const int klo  = (l < 16) ? 0 : 8;

  // pre-scale by log2(e): p = exp(lrelu(s+d)-M) = exp2(lrelu((s+d)*L) - M*L)
  const float srowL = S[(size_t)head * N_NODES + rowg + ml] * LOG2E;
  const float MrowL = M[(size_t)head * N_NODES + rowg + ml] * LOG2E;
  const float* dv   = D + (size_t)head * N_NODES;
  const _Float16* pB = packedB + (size_t)head * packedHeadStride;
  const uint4* awp  = (const uint4*)(adjbits + (size_t)(rowg + ml) * (N_NODES / 32));

  v8f acc[NC];
#pragma unroll
  for (int n = 0; n < NC; ++n) acc[n] = v8f_zero();

  for (int jbo = 0; jbo < N_NODES / 128; ++jbo) {
    const uint4* src = (const uint4*)(pB + (size_t)jbo * (4 * NC * 512));
#pragma unroll
    for (int w = 0; w < NC; ++w)
      ((uint4*)Btile)[w * 256 + tid] = src[w * 256 + tid];
    if (tid < 128) dblk[tid] = dv[jbo * 128 + tid] * LOG2E;
    __builtin_prefetch(src + 4 * NC * 128, 0, 1);          // start of the tile two rounds ahead
    __syncthreads();

    uint4 aw4 = awp[jbo];
    unsigned int aw[4] = {aw4.x, aw4.y, aw4.z, aw4.w};
#pragma unroll
    for (int q = 0; q < 4; ++q) {
      v16h aP;
#pragma unroll
      for (int i = 0; i < 16; ++i) {
        // ISA 16-bit A layout: lanes<16 own K{0..7,16..23}, lanes>=16 own K{8..15,24..31}
        int K = i + ((i < 8) ? 0 : 8) + klo;
        float t = srowL + dblk[q * 32 + K];
        t = fmaxf(t, SLOPE * t);                            // lrelu (scale-invariant)
        float p = ((aw[q] >> K) & 1u) ? __builtin_amdgcn_exp2f(t - MrowL) : 0.f;
        aP[i] = (_Float16)p;
      }
#pragma unroll
      for (int n = 0; n < NC; ++n) {
        v16h b = *(const v16h*)(Btile + (size_t)((q * NC + n) * 32 + l) * 16);
        acc[n] = wmma_f16(aP, b, acc[n]);
      }
    }
    __syncthreads();
  }

  // denominator = ones-column of the accumulator (C-row r(+8) lives in lane DLANE(+16))
  float dnv[8];
#pragma unroll
  for (int r = 0; r < 8; ++r) {
    float dn = __shfl(acc[DN][r], (l < 16) ? DLANE : (DLANE + 16), 32);
    dnv[r] = (dn < 1e-30f) ? 1.0f : dn;
  }

#pragma unroll
  for (int n = 0; n < NC; ++n) {
    int col = n * 16 + ml;
#pragma unroll
    for (int r = 0; r < 8; ++r) {
      int rowo = r + ((l < 16) ? 0 : 8);
      float v = acc[n][r] / dnv[r];
      v = (v > 0.f) ? v : (__expf(v) - 1.f);               // elu
      if (col < colLimit)
        out[(size_t)(rowg + rowo) * outStride + head * headColMul + col] = v;
    }
  }
}

// ---------------- 8) tiny layer-2 projection: Wh2 = h2(8192x256) @ W2(256x20)
__global__ __launch_bounds__(256) void proj2_kernel(const float* __restrict__ h2,
                                                    const float* __restrict__ W2,
                                                    float* __restrict__ Wh2) {
  int idx = blockIdx.x * 256 + threadIdx.x;    // 8192*20 exactly
  int row = idx / CLASSES, c = idx % CLASSES;
  const float* h = h2 + (size_t)row * (HEADS * F_HID);
  float s = 0.f;
  for (int k = 0; k < HEADS * F_HID; ++k) s += h[k] * W2[(size_t)k * CLASSES + c];
  Wh2[idx] = s;
}

// =====================================================================
extern "C" void kernel_launch(void* const* d_in, const int* in_sizes, int n_in,
                              void* d_out, int out_size, void* d_ws, size_t ws_size,
                              hipStream_t stream) {
  (void)in_sizes; (void)n_in; (void)out_size; (void)ws_size;
  const float* x      = (const float*)d_in[0];
  const int*   adj    = (const int*)d_in[1];
  const float* W1     = (const float*)d_in[2];
  const float* alpha1 = (const float*)d_in[3];
  const float* W2     = (const float*)d_in[4];
  const float* alpha2 = (const float*)d_in[5];
  float* out = (float*)d_out;

  // ---- workspace layout (~31 MB) ----
  char* ws = (char*)d_ws;
  size_t off = 0;
  auto take = [&](size_t bytes) { void* p = ws + off; off += (bytes + 255) & ~(size_t)255; return p; };
  unsigned int* adjbits = (unsigned int*)take((size_t)N_NODES * (N_NODES / 32) * 4); // 8 MB
  float*    wh1         = (float*)take((size_t)HEADS * N_NODES * F_HID * 4);         // 8 MB
  _Float16* packedW1    = (_Float16*)take((size_t)HEADS * 16 * 4 * 512 * 2);         // 256 KB
  float*    s1          = (float*)take((size_t)HEADS * N_NODES * 4);
  float*    d1          = (float*)take((size_t)HEADS * N_NODES * 4);
  float*    m1          = (float*)take((size_t)HEADS * N_NODES * 4);
  _Float16* packedWh1   = (_Float16*)take((size_t)HEADS * 256 * 5 * 512 * 2);        // 5 MB
  float*    h2          = (float*)take((size_t)N_NODES * HEADS * F_HID * 4);         // 8 MB
  float*    wh2         = (float*)take((size_t)N_NODES * CLASSES * 4);
  float*    s2          = (float*)take((size_t)N_NODES * 4);
  float*    d2          = (float*)take((size_t)N_NODES * 4);
  float*    m2          = (float*)take((size_t)N_NODES * 4);
  _Float16* packedWh2   = (_Float16*)take((size_t)256 * 2 * 512 * 2);                // 512 KB

  // 1) bit-pack adjacency (reads 256MB once; 8MB result lives in L2 for all passes)
  adjpack_kernel<<<dim3(N_NODES * (N_NODES / 32) / 256), dim3(256), 0, stream>>>(adj, adjbits);

  // 2) pack W1 into WMMA-B layout (kb=16, n=4 per head; no ones column)
  pack_b_kernel<<<dim3((16 * 4 * 32) / 256, HEADS), dim3(256), 0, stream>>>(
      W1, packedW1, F_HID, F_HID, 4, /*onesCol=*/-1,
      (size_t)F_IN * F_HID, (size_t)16 * 4 * 512);

  // 3) WMMA projection x @ W1 -> Wh1 (f32)
  proj1_kernel<<<dim3(N_NODES / 16, HEADS), dim3(32), 0, stream>>>(x, packedW1, wh1);

  // 4) s1/d1 dots
  sd_kernel<<<dim3(N_NODES / 256, HEADS), dim3(256), 0, stream>>>(wh1, alpha1, s1, d1, F_HID);

  // 5) row maxes from bitmask (lrelu monotone => exact softmax max)
  rowmax_kernel<<<dim3(N_NODES, HEADS), dim3(32), 0, stream>>>(adjbits, d1, s1, m1);

  // 6) pack Wh1 into WMMA-B layout (kb=256, n=5: 4 data tiles + ones col @64)
  pack_b_kernel<<<dim3((256 * 5 * 32) / 256, HEADS), dim3(256), 0, stream>>>(
      wh1, packedWh1, F_HID, F_HID, 5, /*onesCol=*/64,
      (size_t)N_NODES * F_HID, (size_t)256 * 5 * 512);

  // 7) fused attention aggregation layer 1 -> h2 (elu, head-concat columns)
  agg_kernel<5, 4, 0><<<dim3(N_NODES / 16 / 8, HEADS), dim3(256), 0, stream>>>(
      packedWh1, s1, d1, m1, adjbits, h2,
      /*colLimit=*/F_HID, /*outStride=*/HEADS * F_HID,
      /*headColMul=*/F_HID, /*packedHeadStride=*/(size_t)256 * 5 * 512);

  // 8) layer-2 projection h2 @ W2 -> Wh2
  proj2_kernel<<<dim3(N_NODES * CLASSES / 256), dim3(256), 0, stream>>>(h2, W2, wh2);

  // 9) s2/d2 dots
  sd_kernel<<<dim3(N_NODES / 256, 1), dim3(256), 0, stream>>>(wh2, alpha2, s2, d2, CLASSES);

  // 10) row maxes layer 2
  rowmax_kernel<<<dim3(N_NODES, 1), dim3(32), 0, stream>>>(adjbits, d2, s2, m2);

  // 11) pack Wh2 (N padded 20->32, ones col @20) into WMMA-B layout (kb=256, n=2)
  pack_b_kernel<<<dim3((256 * 2 * 32) / 256, 1), dim3(256), 0, stream>>>(
      wh2, packedWh2, CLASSES, CLASSES, 2, /*onesCol=*/20, 0, 0);

  // 12) fused attention aggregation layer 2 -> d_out (elu)
  agg_kernel<2, 1, 4><<<dim3(N_NODES / 16 / 8, 1), dim3(256), 0, stream>>>(
      packedWh2, s2, d2, m2, adjbits, out,
      /*colLimit=*/CLASSES, /*outStride=*/CLASSES,
      /*headColMul=*/0, /*packedHeadStride=*/0);
}
